// CrossGATLayer_83889301226229
// MI455X (gfx1250) — compile-verified, matrix operand order checked by reading, not run on hardware
//
#include <hip/hip_runtime.h>
#include <hip/hip_bf16.h>

#define BS   8
#define TT   1024
#define NN   2048      // 2*T
#define FIN  512
#define FOUT 512
#define SK   4         // semantic top-k
#define ALPHA 0.1f

typedef __attribute__((ext_vector_type(16))) __bf16 v16bf;
typedef __attribute__((ext_vector_type(8)))  __bf16 v8bf;
typedef __attribute__((ext_vector_type(8)))  float  v8f;
typedef __attribute__((ext_vector_type(4)))  float  v4f;

// WMMA 16-bit operand fragment addressing (ISA 7.12.2):
//   element (row r, k) of a 16x32 A-tile lives at
//     lane = r + 16*((k>>3)&1),  slot = (k&7) + 8*((k>>4)&1)
// B (32x16, KxN) uses the same map with N in place of r.
// Fragments are stored as [sub][lane][16] contiguous bf16 so each lane's
// operand is one aligned 32-byte span (2x ds_load_b128).

// ---------------------------------------------------------------------------
// Kernel 1: Wh = concat(x_a, x_v) @ W   (bf16 WMMA, f32 accumulate)
// Block tile 128(M) x 64(N); 8 waves, each wave 32x32 (4 accumulators).
// grid = (2048/128, 512/64, BS), block = 256.
// ---------------------------------------------------------------------------
__global__ void __launch_bounds__(256)
gemm_Wh(const float* __restrict__ xa, const float* __restrict__ xv,
        const float* __restrict__ W, float* __restrict__ Wh,
        __bf16* __restrict__ Whb)
{
    const int b   = blockIdx.z;
    const int bm  = blockIdx.x * 128;
    const int bn  = blockIdx.y * 64;
    const int tid = threadIdx.x;
    const int lane = tid & 31, wid = tid >> 5;
    const int mw = wid & 3;        // 0..3 : two M sub-tiles (2mw, 2mw+1)
    const int nw = wid >> 2;       // 0..1 : two N sub-tiles (2nw, 2nw+1)

    __shared__ __bf16 fragA[8 * 32 * 16];   // 8 m-subs
    __shared__ __bf16 fragB[4 * 32 * 16];   // 4 n-subs

    v8f acc[2][2] = {};

    // staging geometry (computed once)
    // A: thread handles chunks c = tid*2, tid*2+1 of 8 elems; 512 chunks = 128x32
    // B: thread handles column n = tid&63, k-octet (tid>>6)*8
    const int nB   = tid & 63;
    const int kqB  = (tid >> 6) << 3;            // 0,8,16,24
    const int laneB = (nB & 15) + 16 * ((kqB >> 3) & 1);
    const int dstB  = ((nB >> 4) * 32 + laneB) * 16 + ((kqB >> 4) & 1) * 8;

    for (int k0 = 0; k0 < FIN; k0 += 32) {
        // ---- stage A (f32 -> bf16, fragment order, b128 in / b128 out) ----
        for (int s = 0; s < 2; ++s) {
            int c  = tid * 2 + s;
            int r  = c >> 2;
            int c0 = (c & 3) << 3;               // 0,8,16,24
            int gm = bm + r;
            const float* src = (gm < TT)
                ? xa + ((size_t)b * TT + gm) * FIN + k0 + c0
                : xv + ((size_t)b * TT + (gm - TT)) * FIN + k0 + c0;
            v4f lo = *(const v4f*)src;
            v4f hi = *(const v4f*)(src + 4);
            v8bf t;
            for (int j = 0; j < 4; ++j) { t[j] = (__bf16)lo[j]; t[4 + j] = (__bf16)hi[j]; }
            int laneA = (r & 15) + 16 * ((c0 >> 3) & 1);
            int dstA  = ((r >> 4) * 32 + laneA) * 16 + ((c0 >> 4) & 1) * 8;
            *(v8bf*)&fragA[dstA] = t;
        }
        // ---- stage B: 8 k-strided f32 at fixed n (coalesced across lanes) ----
        {
            const float* src = W + (size_t)(k0 + kqB) * FOUT + bn + nB;
            v8bf t;
            for (int q = 0; q < 8; ++q) t[q] = (__bf16)src[(size_t)q * FOUT];
            *(v8bf*)&fragB[dstB] = t;
            if (k0 + 32 < FIN)
                __builtin_prefetch(src + (size_t)32 * FOUT, 0, 3);
        }
        __syncthreads();

        v16bf af[2], bfr[2];
        for (int im = 0; im < 2; ++im)
            af[im] = *(const v16bf*)&fragA[((mw * 2 + im) * 32 + lane) * 16];
        for (int in = 0; in < 2; ++in)
            bfr[in] = *(const v16bf*)&fragB[((nw * 2 + in) * 32 + lane) * 16];

        for (int im = 0; im < 2; ++im)
            for (int in = 0; in < 2; ++in)
                acc[im][in] = __builtin_amdgcn_wmma_f32_16x16x32_bf16(
                    false, af[im], false, bfr[in], (short)0, acc[im][in], false, false);
        __syncthreads();
    }

    // C/D layout: VGPR r -> M = r + 8*(lane>>4); N = lane&15
    const int moff = ((lane >> 4) << 3);
    const int ncol = (lane & 15);
    for (int im = 0; im < 2; ++im) {
        for (int in = 0; in < 2; ++in) {
            int gm0 = bm + (mw * 2 + im) * 16 + moff;
            int gn  = bn + (nw * 2 + in) * 16 + ncol;
            for (int r = 0; r < 8; ++r) {
                size_t o = ((size_t)b * NN + gm0 + r) * FOUT + gn;
                float v = acc[im][in][r];
                Wh[o]  = v;
                Whb[o] = (__bf16)v;
            }
        }
    }
}

// ---------------------------------------------------------------------------
// Kernel 2: per-row Wh1 = Wh.a1, Wh2 = Wh.a2, norms = |Wh|^2
// One wave32 per row; grid.x = BS*NN/8, block = 256 (8 waves).
// ---------------------------------------------------------------------------
__global__ void __launch_bounds__(256)
row_stats(const float* __restrict__ Wh, const float* __restrict__ a,
          float* __restrict__ Wh1, float* __restrict__ Wh2,
          float* __restrict__ norms)
{
    const int row  = blockIdx.x * 8 + (threadIdx.x >> 5);
    const int lane = threadIdx.x & 31;
    const float* p = Wh + (size_t)row * FOUT;
    float s1 = 0.f, s2 = 0.f, nn = 0.f;
    for (int k = lane; k < FOUT; k += 32) {
        float w = p[k];
        s1 += w * a[k];
        s2 += w * a[FOUT + k];
        nn += w * w;
    }
    for (int off = 16; off; off >>= 1) {
        s1 += __shfl_xor(s1, off);
        s2 += __shfl_xor(s2, off);
        nn += __shfl_xor(nn, off);
    }
    if (lane == 0) { Wh1[row] = s1; Wh2[row] = s2; norms[row] = nn; }
}

// ---------------------------------------------------------------------------
// Kernel 3: semantic top-4 inside aa / vv blocks via WMMA Gram tiles.
// pd[i,j] = 2<xi,xj> - |xi|^2 - |xj|^2. The block's full A operand
// (32 rows x 512 K = 16 fragments/wave) is held in registers; inner loop is
// 2x ds_load_b128 (B fragment) + 1 WMMA. Fused streaming top-4.
// grid = (1024/32, 2 sides, BS), block = 256 (waves: 2 in M, 4 in N).
// ---------------------------------------------------------------------------
__global__ void __launch_bounds__(256)
semantic_topk(const __bf16* __restrict__ Whb, const float* __restrict__ norms,
              int* __restrict__ topk)
{
    const int b = blockIdx.z, side = blockIdx.y;
    const int row0 = blockIdx.x * 32;
    const int tid = threadIdx.x, lane = tid & 31, wid = tid >> 5;
    const int mw = wid & 1;     // 0..1 : M sub-tile
    const int nw = wid >> 1;    // 0..3 : N sub-tile

    __shared__ __bf16 fragA[16 * 2 * 32 * 16];  // [ksec][msub][lane][16] = 32 KiB
    __shared__ __bf16 fragB[4 * 32 * 16];       // [nsub][lane][16] for one ksec
    __shared__ float  ldsPD[32][64 + 1];
    __shared__ float  mrgV[32][32];
    __shared__ int    mrgI[32][32];

    const size_t rowbase = (size_t)b * NN + (size_t)side * TT;

    // ---- stage A in fragment order: 32x512 bf16, b128 in / b128 out ----
    for (int s = 0; s < 8; ++s) {
        int chunk = tid + 256 * s;               // 2048 chunks of 8
        int r    = chunk >> 6;                   // 0..31
        int kc8  = (chunk & 63) << 3;            // 0..504
        int ksec = kc8 >> 5;
        int kk0  = kc8 & 31;                     // 0,8,16,24
        v8bf t = *(const v8bf*)&Whb[(rowbase + row0 + r) * FOUT + kc8];
        int laneA = (r & 15) + 16 * ((kk0 >> 3) & 1);
        int dst = (((ksec * 2 + (r >> 4)) * 32) + laneA) * 16 + ((kk0 >> 4) & 1) * 8;
        *(v8bf*)&fragA[dst] = t;
    }
    __syncthreads();

    // A fragments -> registers (full K, this wave's m sub-tile)
    v16bf af[16];
#pragma unroll
    for (int ks = 0; ks < 16; ++ks)
        af[ks] = *(const v16bf*)&fragA[((ks * 2 + mw) * 32 + lane) * 16];

    // per-thread streaming top-4 (8 threads per row, 8 cols each per tile)
    float tv[SK] = { -3e38f, -3e38f, -3e38f, -3e38f };
    int   ti[SK] = { -1, -1, -1, -1 };
    const int prow  = tid >> 3;
    const int pslot = tid & 7;

    // B staging geometry: thread -> (col j = tid>>2, k-octet (tid&3)*8)
    const int jB   = tid >> 2;                   // 0..63
    const int kk0B = (tid & 3) << 3;             // 0,8,16,24
    const int laneBv = (jB & 15) + 16 * ((kk0B >> 3) & 1);
    const int dstB   = ((jB >> 4) * 32 + laneBv) * 16 + ((kk0B >> 4) & 1) * 8;

    for (int ct = 0; ct < 16; ++ct) {
        const int col0 = ct * 64;
        v8f acc = {};
#pragma unroll 4
        for (int ksec = 0; ksec < 16; ++ksec) {
            // stage B fragment slab for this ksec (one b128 in, one b128 out)
            *(v8bf*)&fragB[dstB] =
                *(const v8bf*)&Whb[(rowbase + col0 + jB) * FOUT + ksec * 32 + kk0B];
            __syncthreads();
            v16bf bfr = *(const v16bf*)&fragB[(nw * 32 + lane) * 16];
            acc = __builtin_amdgcn_wmma_f32_16x16x32_bf16(
                false, af[ksec], false, bfr, (short)0, acc, false, false);
            __syncthreads();
        }

        // pd tile -> LDS
        {
            const int m = mw * 16 + ((lane >> 4) << 3);
            const int n = nw * 16 + (lane & 15);
            const float njv = norms[rowbase + col0 + n];
            for (int r = 0; r < 8; ++r) {
                float niv = norms[rowbase + row0 + m + r];
                ldsPD[m + r][n] = 2.0f * acc[r] - niv - njv;
            }
        }
        __syncthreads();

        // streaming top-4 update
        for (int q = 0; q < 8; ++q) {
            int c = pslot * 8 + q;
            float v = ldsPD[prow][c];
            if (v > tv[SK - 1]) {
                int p = SK - 1;
                while (p > 0 && v > tv[p - 1]) {
                    tv[p] = tv[p - 1]; ti[p] = ti[p - 1]; --p;
                }
                tv[p] = v; ti[p] = col0 + c;
            }
        }
        __syncthreads();
    }

    // merge 8 partial top-4 lists per row -> final top-4
    for (int q = 0; q < SK; ++q) {
        mrgV[prow][pslot * SK + q] = tv[q];
        mrgI[prow][pslot * SK + q] = ti[q];
    }
    __syncthreads();
    if (tid < 32) {
        const int row = tid;
        int* op = topk + (((size_t)(b * 2 + side) * TT) + row0 + row) * SK;
        for (int s = 0; s < SK; ++s) {
            float best = -3e38f; int bi = -1, bp = -1;
            for (int u = 0; u < 32; ++u) {
                float v = mrgV[row][u];
                if (v > best) { best = v; bi = mrgI[row][u]; bp = u; }
            }
            if (bp >= 0) mrgV[row][bp] = -3e38f;
            op[s] = bi;
        }
    }
}

// ---------------------------------------------------------------------------
// Kernel 4: sparse attention + aggregate + elu.
// Each node has <= 7 neighbors (temporal band of 3 U semantic top-4, dedup'd).
// grid = (NN, BS), block = 128 (4 features / thread).
// ---------------------------------------------------------------------------
__global__ void __launch_bounds__(128)
attn_aggregate(const float* __restrict__ Wh, const float* __restrict__ Wh1,
               const float* __restrict__ Wh2, const int* __restrict__ topk,
               float* __restrict__ out)
{
    const int i = blockIdx.x;             // 0..2047
    const int b = blockIdx.y;
    const int side = (i >= TT) ? 1 : 0;
    const int li   = i - side * TT;
    const size_t rb = (size_t)b * NN;

    int cand[3 + SK]; int nc = 0;
    for (int d = -1; d <= 1; ++d) {
        int lj = li + d;
        if (lj >= 0 && lj < TT) cand[nc++] = side * TT + lj;
    }
    const int* tk = topk + (((size_t)(b * 2 + side) * TT) + li) * SK;
    for (int q = 0; q < SK; ++q) {
        int j = tk[q];
        if (j < 0) continue;
        j += side * TT;
        bool dup = false;
        for (int u = 0; u < nc; ++u) if (cand[u] == j) dup = true;
        if (!dup) cand[nc++] = j;
    }

    // additive score + leaky relu + softmax over the neighbor set
    float wv[3 + SK];
    const float s1 = Wh1[rb + i];
    float emax = -3e38f;
    for (int u = 0; u < nc; ++u) {
        float e = s1 + Wh2[rb + cand[u]];
        e = (e > 0.f) ? e : ALPHA * e;
        wv[u] = e;
        emax = fmaxf(emax, e);
    }
    float den = 0.f;
    for (int u = 0; u < nc; ++u) { wv[u] = __expf(wv[u] - emax); den += wv[u]; }
    const float inv = 1.0f / den;

    const int f0 = threadIdx.x * 4;
    float acc[4] = { 0.f, 0.f, 0.f, 0.f };
    for (int u = 0; u < nc; ++u) {
        const float* src = Wh + (rb + cand[u]) * FOUT + f0;
        const float w = wv[u] * inv;
        for (int q = 0; q < 4; ++q) acc[q] += w * src[q];
    }
    // outputs: h_a flat (BS,T,F) then h_v flat (BS,T,F)
    float* op = out + (side ? (size_t)BS * TT * FOUT : 0)
                    + ((size_t)b * TT + li) * FOUT + f0;
    for (int q = 0; q < 4; ++q) {
        float v = acc[q];
        op[q] = (v > 0.f) ? v : (__expf(v) - 1.0f);
    }
}

// ---------------------------------------------------------------------------
extern "C" void kernel_launch(void* const* d_in, const int* in_sizes, int n_in,
                              void* d_out, int out_size, void* d_ws, size_t ws_size,
                              hipStream_t stream)
{
    const float* xa = (const float*)d_in[0];   // (8,1024,512)
    const float* xv = (const float*)d_in[1];   // (8,1024,512)
    const float* W  = (const float*)d_in[2];   // (512,512)
    const float* a  = (const float*)d_in[3];   // (1024,1)
    float* out = (float*)d_out;                // h_a then h_v, 8*2048*512 floats

    // workspace layout
    char* ws = (char*)d_ws;
    const size_t WH_BYTES  = (size_t)BS * NN * FOUT * sizeof(float);     // 32 MiB
    const size_t WHB_BYTES = (size_t)BS * NN * FOUT * sizeof(__bf16);    // 16 MiB
    float*  Wh    = (float*)ws;
    __bf16* Whb   = (__bf16*)(ws + WH_BYTES);
    float*  Wh1   = (float*)(ws + WH_BYTES + WHB_BYTES);
    float*  Wh2   = Wh1 + (size_t)BS * NN;
    float*  norms = Wh2 + (size_t)BS * NN;
    int*    topk  = (int*)(norms + (size_t)BS * NN);

    gemm_Wh<<<dim3(NN / 128, FOUT / 64, BS), 256, 0, stream>>>(xa, xv, W, Wh, Whb);

    row_stats<<<dim3(BS * NN / 8), 256, 0, stream>>>(Wh, a, Wh1, Wh2, norms);

    semantic_topk<<<dim3(TT / 32, 2, BS), 256, 0, stream>>>(Whb, norms, topk);

    attn_aggregate<<<dim3(NN, BS), 128, 0, stream>>>(Wh, Wh1, Wh2, topk, out);
}
